// MultiHeadAttention_30416958390733
// MI455X (gfx1250) — matrix operand order repaired and checked by reading, not compiled, not run on hardware
//
#include <hip/hip_runtime.h>
#include <hip/hip_bf16.h>

// ---------------------------------------------------------------------------
// MHA with projections, bf16 WMMA pipeline for gfx1250 (MI455X).
//   B=2, S=2048, LATENT=1024, HEADS=16, DEPTH=64
// Stage 0: additive padding-mask precompute (f32)
// Stage 1: proj GEMM, 32x64 register-blocked per wave, v_wmma_f32_16x16x32_bf16
// Stage 2: flash attention; workgroup = 8 q-tiles of one (b,h); K/V chunks
//          staged in LDS via async-load-to-LDS (ASYNCcnt) when available.
// ---------------------------------------------------------------------------

#define BATCH  2
#define SEQ    2048
#define LATENT 1024
#define NHEADS 16
#define DEPTH  64

typedef __attribute__((ext_vector_type(16))) __bf16 v16bf;
typedef __attribute__((ext_vector_type(8)))  __bf16 v8bf;
typedef __attribute__((ext_vector_type(8)))  float  v8f;

#if defined(__AMDGCN__) && __has_builtin(__builtin_amdgcn_global_load_async_to_lds_b128)
#define HAVE_ASYNC_LDS 1
#else
#define HAVE_ASYNC_LDS 0
#endif

// Builtin parameter type (from hipcc diagnostic):
//   int __attribute__((vector_size(16))) __device__ *  (AS1 src)
typedef int vint4_b128 __attribute__((__vector_size__(16)));
#define AS1_B128(p) ((__attribute__((address_space(1))) vint4_b128*)(p))
#define AS3_B128(p) ((__attribute__((address_space(3))) vint4_b128*)(p))

__device__ __forceinline__ void wait_asynccnt0() {
#if defined(__AMDGCN__)
#if __has_builtin(__builtin_amdgcn_s_wait_asynccnt)
    __builtin_amdgcn_s_wait_asynccnt(0);
#else
    asm volatile("s_wait_asynccnt 0x0" ::: "memory");
#endif
#endif
}

// ---------------------------------------------------------------------------
// Stage 0: negadd[b*S+s] = (1 - v_mask) * 1e9  (additive padding penalty)
// ---------------------------------------------------------------------------
__global__ __launch_bounds__(256) void mask_init_kernel(
    const int* __restrict__ vm, float* __restrict__ negadd, int n)
{
    const int i = blockIdx.x * 256 + threadIdx.x;
    if (i < n) negadd[i] = (1.0f - (float)vm[i]) * 1.0e9f;
}

// ---------------------------------------------------------------------------
// Stage 1: OUT[b][h][s][d] (bf16) = X[b*S+s][:] @ W[:][h*D+d]
// One wave computes a 32x64 output block (2 M-tiles x 4 N-tiles):
// per K-step: 2 A-frags + 4 B-frags -> 8 WMMAs.
// ---------------------------------------------------------------------------
__global__ __launch_bounds__(256) void proj_bf16_kernel(
    const float* __restrict__ X, const float* __restrict__ W,
    __bf16* __restrict__ OUT)
{
    const int lane = threadIdx.x & 31;
    const int wv   = threadIdx.x >> 5;
    const int w    = blockIdx.x * 8 + wv;
    const int tilesN = LATENT / 64;              // 16 blocks of 64 cols
    const int m0 = (w / tilesN) * 32;
    const int n0 = (w % tilesN) * 64;
    const int hf  = lane >> 4;
    const int l15 = lane & 15;
    const int kb  = hf * 8;

    v8f acc[2][4] = {{{}, {}, {}, {}}, {{}, {}, {}, {}}};

    for (int k0 = 0; k0 < LATENT; k0 += 32) {
        // --- A fragments: rows m0+mi*16 .. +15, K = k0..k0+31 ---
        v16bf a[2];
#pragma unroll
        for (int mi = 0; mi < 2; ++mi) {
            const float* ap = X + (size_t)(m0 + mi * 16 + l15) * LATENT + k0 + kb;
            __builtin_prefetch(ap + 32, 0, 0);   // next K-step
            float af[16];
            *(float4*)(af + 0)  = *(const float4*)(ap + 0);
            *(float4*)(af + 4)  = *(const float4*)(ap + 4);
            *(float4*)(af + 8)  = *(const float4*)(ap + 16);
            *(float4*)(af + 12) = *(const float4*)(ap + 20);
#pragma unroll
            for (int i = 0; i < 16; ++i) a[mi][i] = (__bf16)af[i];
        }
        // --- B fragments: W rows k0..k0+31, cols n0+ni*16 .. +15 ---
        const int kRow = k0 + kb;
        const float* bp = W + (size_t)kRow * LATENT + n0 + l15;
        __builtin_prefetch(bp + (size_t)32 * LATENT, 0, 0);  // next K-step
        v16bf bfr[4];
#pragma unroll
        for (int ni = 0; ni < 4; ++ni) {
            float bfl[16];
#pragma unroll
            for (int i = 0; i < 16; ++i)
                bfl[i] = bp[(size_t)((i & 7) + (i >> 3) * 16) * LATENT + ni * 16];
#pragma unroll
            for (int i = 0; i < 16; ++i) bfr[ni][i] = (__bf16)bfl[i];
        }
#pragma unroll
        for (int mi = 0; mi < 2; ++mi)
#pragma unroll
            for (int ni = 0; ni < 4; ++ni)
                acc[mi][ni] = __builtin_amdgcn_wmma_f32_16x16x32_bf16(
                    false, a[mi], false, bfr[ni], (short)0, acc[mi][ni],
                    false, false);
    }

    // C layout: VGPR r -> M = hf*8 + r ; N = l15.  Store to [b][h][s][d] bf16.
#pragma unroll
    for (int ni = 0; ni < 4; ++ni) {
        const int col = n0 + ni * 16 + l15;
        const int hh  = col >> 6;
        const int dd  = col & 63;
#pragma unroll
        for (int mi = 0; mi < 2; ++mi) {
#pragma unroll
            for (int r = 0; r < 8; ++r) {
                const int row = m0 + mi * 16 + hf * 8 + r;
                const int bb  = row >> 11;
                const int ss  = row & (SEQ - 1);
                OUT[(((size_t)bb * NHEADS + hh) * SEQ + ss) * DEPTH + dd] =
                    (__bf16)acc[mi][ni][r];
            }
        }
    }
}

// ---------------------------------------------------------------------------
// Stage 2: flash attention.
// Workgroup (256 thr, 8 waves) owns 8 consecutive q-tiles of one (b,h).
// Per 32-key chunk: K/V (2 x 4KB, contiguous) staged into LDS cooperatively
// (async-to-LDS when available), fragments read back with ds_load_b128.
// Scores computed transposed: S^T C-layout == P A-fragment layout in-lane.
// ---------------------------------------------------------------------------
__global__ __launch_bounds__(256) void attn_flash_kernel(
    const __bf16* __restrict__ qh, const __bf16* __restrict__ kh,
    const __bf16* __restrict__ vh, const float* __restrict__ negadd,
    float* __restrict__ out)
{
    const int tid  = threadIdx.x;
    const int lane = tid & 31;
    const int hf   = lane >> 4;
    const int l15  = lane & 15;
    const int wv   = tid >> 5;

    const int blk  = blockIdx.x;                 // 0..511
    const int qblk = blk & (SEQ / 128 - 1);      // 16 q-blocks per (b,h)
    const int bh   = blk >> 4;                   // b*NHEADS + h
    const int b    = bh >> 4;
    const int h    = bh & (NHEADS - 1);
    const int q0   = qblk * 128 + wv * 16;       // this wave's q-tile

    const size_t base = (size_t)bh * SEQ * DEPTH;

    __shared__ __bf16 kTile[32 * DEPTH];         // 4 KB
    __shared__ __bf16 vTile[32 * DEPTH];         // 4 KB

    // Q^T B-fragments (loaded once): lane = query col, elements = d rows (table order).
    v16bf bq[2];
#pragma unroll
    for (int dc = 0; dc < 2; ++dc) {
        const int d = dc * 32 + hf * 8;
#pragma unroll
        for (int j = 0; j < 16; ++j)
            bq[dc][j] = qh[base + (size_t)(q0 + l15) * DEPTH + d + (j & 7) + (j >> 3) * 16];
    }

    float m_run = -3.0e38f, l_run = 0.0f;
    v8f acc[4] = {{}, {}, {}, {}};

    const int qEnd   = q0 + 15;
    const int blkEnd = qblk * 128 + 127;         // workgroup's causal bound
    for (int kv0 = 0; kv0 <= blkEnd; kv0 += 32) {
        // ---- cooperative stage of contiguous 4KB K and V chunks ----
        const __bf16* kg = kh + base + (size_t)kv0 * DEPTH;
        const __bf16* vg = vh + base + (size_t)kv0 * DEPTH;
#if HAVE_ASYNC_LDS
        __builtin_amdgcn_global_load_async_to_lds_b128(
            AS1_B128(kg + tid * 8), AS3_B128(&kTile[tid * 8]), 0, 0);
        __builtin_amdgcn_global_load_async_to_lds_b128(
            AS1_B128(vg + tid * 8), AS3_B128(&vTile[tid * 8]), 0, 0);
        wait_asynccnt0();
#else
        *(v8bf*)(kTile + tid * 8) = *(const v8bf*)(kg + tid * 8);
        *(v8bf*)(vTile + tid * 8) = *(const v8bf*)(vg + tid * 8);
#endif
        __syncthreads();

        if (kv0 <= qEnd) {                       // wave-uniform causal skip
            // ---- S^T tiles from LDS K ----
            v8f st[2];
#pragma unroll
            for (int t = 0; t < 2; ++t) {
                const int krow = t * 16 + l15;   // A: lane = M = key
                v8f ct = {};
#pragma unroll
                for (int dc = 0; dc < 2; ++dc) {
                    const __bf16* kp = kTile + krow * DEPTH + dc * 32 + hf * 8;
                    v8bf klo = *(const v8bf*)(kp);
                    v8bf khi = *(const v8bf*)(kp + 16);
                    v16bf ak;
#pragma unroll
                    for (int j = 0; j < 8; ++j) { ak[j] = klo[j]; ak[8 + j] = khi[j]; }
                    ct = __builtin_amdgcn_wmma_f32_16x16x32_bf16(
                            false, ak, false, bq[dc], (short)0, ct, false, false);
                }
                st[t] = ct;
            }

            // ---- masks + online softmax (query = q0 + l15 per lane) ----
            const int qi = q0 + l15;
            float p[16];
            float mloc = -3.0e38f;
#pragma unroll
            for (int t = 0; t < 2; ++t) {
                const float* np = negadd + b * SEQ + kv0 + t * 16 + hf * 8;
                float nmf[8];
                *(float4*)(nmf + 0) = *(const float4*)(np + 0);
                *(float4*)(nmf + 4) = *(const float4*)(np + 4);
#pragma unroll
                for (int r = 0; r < 8; ++r) {
                    const int key = kv0 + t * 16 + hf * 8 + r;
                    float s = st[t][r] * 0.125f - nmf[r];      // scale + padding
                    if (key > qi) s -= 1.0e9f;                 // causal
                    p[t * 8 + r] = s;
                    mloc = fmaxf(mloc, s);
                }
            }
            mloc = fmaxf(mloc, __shfl_xor(mloc, 16, 32));
            const float m_new = fmaxf(m_run, mloc);
            const float corr  = __expf(m_run - m_new);
            float rsum = 0.0f;
#pragma unroll
            for (int i = 0; i < 16; ++i) { p[i] = __expf(p[i] - m_new); rsum += p[i]; }
            rsum += __shfl_xor(rsum, 16, 32);
            l_run = l_run * corr + rsum;
            m_run = m_new;

            float corr8[8];
#pragma unroll
            for (int r = 0; r < 8; ++r)
                corr8[r] = __shfl(corr, ((lane & 16) >> 1) + r, 32);
#pragma unroll
            for (int t = 0; t < 4; ++t)
#pragma unroll
                for (int r = 0; r < 8; ++r) acc[t][r] *= corr8[r];

            // ---- P fragment (in-lane) + O += P x V from LDS ----
            v16bf pa;
#pragma unroll
            for (int i = 0; i < 16; ++i) pa[i] = (__bf16)p[i];
            const int vrow = hf * 8;             // B: lane = N = d col, elements = kv rows (table order)
            const __bf16* vp = vTile + vrow * DEPTH + l15;
#pragma unroll
            for (int t = 0; t < 4; ++t) {
                v16bf bv;
#pragma unroll
                for (int j = 0; j < 16; ++j) bv[j] = vp[((j & 7) + (j >> 3) * 16) * DEPTH + t * 16];
                acc[t] = __builtin_amdgcn_wmma_f32_16x16x32_bf16(
                            false, pa, false, bv, (short)0, acc[t], false, false);
            }
        }
        __syncthreads();                         // LDS safe to overwrite
    }

    // ---- normalize and store: out[b][q][h*64 + d], f32 ----
    float l8[8];
#pragma unroll
    for (int r = 0; r < 8; ++r)
        l8[r] = __shfl(l_run, ((lane & 16) >> 1) + r, 32);
#pragma unroll
    for (int t = 0; t < 4; ++t) {
#pragma unroll
        for (int r = 0; r < 8; ++r) {
            const int qq = q0 + hf * 8 + r;
            const int dd = t * 16 + l15;
            out[((size_t)b * SEQ + qq) * LATENT + h * DEPTH + dd] =
                acc[t][r] / l8[r];
        }
    }
}

// ---------------------------------------------------------------------------
extern "C" void kernel_launch(void* const* d_in, const int* in_sizes, int n_in,
                              void* d_out, int out_size, void* d_ws, size_t ws_size,
                              hipStream_t stream) {
    const float* q     = (const float*)d_in[0];
    const float* k     = (const float*)d_in[1];
    const float* v     = (const float*)d_in[2];
    const int*   vmask = (const int*)  d_in[3];
    const float* Wq    = (const float*)d_in[4];
    const float* Wk    = (const float*)d_in[5];
    const float* Wv    = (const float*)d_in[6];
    float* out = (float*)d_out;

    // Workspace: 3 bf16 tensors [B][H][S][D] (3 x 8MB) + f32 additive mask
    const size_t sec = (size_t)BATCH * NHEADS * SEQ * DEPTH;
    __bf16* qh = (__bf16*)d_ws;
    __bf16* kh = qh + sec;
    __bf16* vh = kh + sec;
    float* negadd = (float*)(vh + sec);

    mask_init_kernel<<<(BATCH * SEQ) / 256, 256, 0, stream>>>(
        vmask, negadd, BATCH * SEQ);

    // Projections: (B*S/32) * (LATENT/64) waves, 8 waves/block
    const int projBlocks = ((BATCH * SEQ / 32) * (LATENT / 64)) / 8;   // 256
    proj_bf16_kernel<<<projBlocks, 256, 0, stream>>>(q, Wq, qh);
    proj_bf16_kernel<<<projBlocks, 256, 0, stream>>>(k, Wk, kh);
    proj_bf16_kernel<<<projBlocks, 256, 0, stream>>>(v, Wv, vh);

    // Attention: B*H*(S/128) workgroups of 256 threads
    const int attnBlocks = BATCH * NHEADS * (SEQ / 128);               // 512
    attn_flash_kernel<<<attnBlocks, 256, 0, stream>>>(qh, kh, vh, negadd, out);
}
